// torchmodule_GAT_role_classif_43679817400823
// MI455X (gfx1250) — matrix operand order repaired and checked by reading, not compiled, not run on hardware
//
#include <hip/hip_runtime.h>
#include <stdint.h>
#include <stddef.h>

// ---------------------------------------------------------------------------
// Problem constants (from the reference)
// ---------------------------------------------------------------------------
#define NNODE 20000
#define DIN   256
#define H1    128
#define RNG   16
#define NHEAD 4
#define DHEAD 32
#define HC    128          // == H1 (residual requirement)
#define NCLS  32
#define NEDGE 320000
#define NEP   (NEDGE + NNODE)   // edges + self loops = 340000

typedef __attribute__((ext_vector_type(16))) __bf16 v16bf;
typedef __attribute__((ext_vector_type(8)))  float  v8f;

union BFrag { v16bf v; uint4 q[2]; };

// ---------------------------------------------------------------------------
// Helpers
// ---------------------------------------------------------------------------
__device__ __forceinline__ unsigned short f2bf(float f) {
  unsigned int u = __float_as_uint(f);
  u += 0x7FFFu + ((u >> 16) & 1u);      // round-to-nearest-even
  return (unsigned short)(u >> 16);
}
__device__ __forceinline__ float bf2f(unsigned short h) {
  return __uint_as_float(((unsigned int)h) << 16);
}
__device__ __forceinline__ v8f wmma_bf16(const BFrag& a, const BFrag& b, v8f c) {
  return __builtin_amdgcn_wmma_f32_16x16x32_bf16(false, a.v, false, b.v,
                                                 (short)0, c, false, false);
}
// Native CDNA5 float atomics (device scope so later kernels see results)
__device__ __forceinline__ void atomic_add_f32(float* p, float v) {
  asm volatile("global_atomic_add_f32 %0, %1, off scope:SCOPE_DEV"
               :: "v"(p), "v"(v) : "memory");
}
__device__ __forceinline__ void atomic_max_f32(float* p, float v) {
  asm volatile("global_atomic_max_num_f32 %0, %1, off scope:SCOPE_DEV"
               :: "v"(p), "v"(v) : "memory");
}

// ---------------------------------------------------------------------------
// Prep kernels: bf16 hi/lo splits of weights, Yb = einsum("hrd,hd->hr")
// ---------------------------------------------------------------------------
__global__ void k_prep_wsplit(const float* __restrict__ W,
                              unsigned short* __restrict__ hi,
                              unsigned short* __restrict__ lo) {
  int i = blockIdx.x * 256 + threadIdx.x;
  if (i >= H1 * RNG * DIN) return;
  float x = W[i];
  unsigned short h = f2bf(x);
  hi[i] = h;
  lo[i] = f2bf(x - bf2f(h));
}

__global__ void k_prep_yb(const float* __restrict__ W,
                          const float* __restrict__ bvec,
                          float* __restrict__ Yb) {
  int m = blockIdx.x * 256 + threadIdx.x;
  if (m >= H1 * RNG) return;
  int hh = m >> 4;
  const float* wrow = W + (size_t)m * DIN;
  const float* bb = bvec + (size_t)hh * DIN;
  float s = 0.f;
  for (int d = 0; d < DIN; ++d) s += wrow[d] * bb[d];
  Yb[m] = s;
}

// src[k*J + j] (f32) -> hi/lo[j*K + k] (bf16 split, transposed for B frags)
__global__ void k_prep_ct(const float* __restrict__ src,
                          unsigned short* __restrict__ hi,
                          unsigned short* __restrict__ lo, int K, int J) {
  int i = blockIdx.x * 256 + threadIdx.x;
  if (i >= K * J) return;
  int k = i / J, j = i % J;
  float x = src[i];
  unsigned short h = f2bf(x);
  hi[(size_t)j * K + k] = h;
  lo[(size_t)j * K + k] = f2bf(x - bf2f(h));
}

// ---------------------------------------------------------------------------
// Bilinear head: H[n,h] = sum_r diag[h,r]*(W[h,r,:]·X0 - c)*(W[h,r,:]·X1 - c) + b0
// GEMM tile [16 r x 16 cols] via split-bf16 WMMA; reduction fused in registers.
// One workgroup = 16 nodes (32 node-channel columns), 8 waves sweep h (2 at a
// time).  B fragments are re-read from LDS each iteration (opaque offset stops
// LICM from hoisting them into spilled registers).
// ---------------------------------------------------------------------------
#define XKS 264   // LDS K stride (256 + 8 pad, keeps 16B alignment)

__global__ void k_bilinear(const float* __restrict__ X,
                           const unsigned short* __restrict__ Whi,
                           const unsigned short* __restrict__ Wlo,
                           const float* __restrict__ diag,
                           const float* __restrict__ Yb,
                           const float* __restrict__ bias0,
                           float* __restrict__ Hout) {
  __shared__ __align__(16) unsigned short sXhi[32 * XKS];
  __shared__ __align__(16) unsigned short sXlo[32 * XKS];

  const int nodeBase = blockIdx.x * 16;

  // Cooperative load of X tile [16 nodes][256 d][2 e] -> LDS [col=e*16+n][k]
  for (int i = threadIdx.x; i < 16 * 512; i += 256) {
    int node = i >> 9;
    int rem  = i & 511;
    int d    = rem >> 1;
    int e    = rem & 1;
    float x  = X[(size_t)(nodeBase + node) * 512 + rem];
    int col  = e * 16 + node;
    unsigned short h = f2bf(x);
    sXhi[col * XKS + d] = h;
    sXlo[col * XKS + d] = f2bf(x - bf2f(h));
  }
  __syncthreads();

  const int w = threadIdx.x >> 5;
  const int ln = threadIdx.x & 31;
  const int lane15 = ln & 15;
  const int hlf = ln >> 4;

  for (int hb = w * 2; hb < H1; hb += 16) {
    const int h0 = hb, h1 = hb + 1;
    v8f acc00 = {}, acc01 = {};   // h0: e=0 / e=1 columns
    v8f acc10 = {}, acc11 = {};   // h1
#pragma unroll 1
    for (int ks = 0; ks < 8; ++ks) {
      int zz = 0;
      asm volatile("" : "+v"(zz));   // opaque 0: keep B loads inside the loop
      BFrag a0Hi, a0Lo, a1Hi, a1Lo, bHi, bLo;
      // A fragments for the two h rows (L2-resident pre-split weights)
      const unsigned short* a0h = Whi + (size_t)(h0 * RNG + lane15) * DIN + ks * 32;
      a0Hi.q[0] = *(const uint4*)(a0h + hlf * 8);
      a0Hi.q[1] = *(const uint4*)(a0h + 16 + hlf * 8);
      const unsigned short* a0l = Wlo + (size_t)(h0 * RNG + lane15) * DIN + ks * 32;
      a0Lo.q[0] = *(const uint4*)(a0l + hlf * 8);
      a0Lo.q[1] = *(const uint4*)(a0l + 16 + hlf * 8);
      const unsigned short* a1h = Whi + (size_t)(h1 * RNG + lane15) * DIN + ks * 32;
      a1Hi.q[0] = *(const uint4*)(a1h + hlf * 8);
      a1Hi.q[1] = *(const uint4*)(a1h + 16 + hlf * 8);
      const unsigned short* a1l = Wlo + (size_t)(h1 * RNG + lane15) * DIN + ks * 32;
      a1Lo.q[0] = *(const uint4*)(a1l + hlf * 8);
      a1Lo.q[1] = *(const uint4*)(a1l + 16 + hlf * 8);
#pragma unroll
      for (int ct = 0; ct < 2; ++ct) {
        const unsigned short* bh =
            sXhi + zz + (ct * 16 + lane15) * XKS + ks * 32;
        bHi.q[0] = *(const uint4*)(bh + hlf * 8);
        bHi.q[1] = *(const uint4*)(bh + 16 + hlf * 8);
        const unsigned short* blp =
            sXlo + zz + (ct * 16 + lane15) * XKS + ks * 32;
        bLo.q[0] = *(const uint4*)(blp + hlf * 8);
        bLo.q[1] = *(const uint4*)(blp + 16 + hlf * 8);
        v8f a = (ct == 0) ? acc00 : acc01;
        a = wmma_bf16(a0Hi, bHi, a);
        a = wmma_bf16(a0Hi, bLo, a);
        a = wmma_bf16(a0Lo, bHi, a);
        if (ct == 0) acc00 = a; else acc01 = a;
        v8f b = (ct == 0) ? acc10 : acc11;
        b = wmma_bf16(a1Hi, bHi, b);
        b = wmma_bf16(a1Hi, bLo, b);
        b = wmma_bf16(a1Lo, bHi, b);
        if (ct == 0) acc10 = b; else acc11 = b;
      }
    }
    // Fused reduction over r: lane holds rows r = v + 8*hlf for column lane15
    float s0 = 0.f, s1 = 0.f;
#pragma unroll
    for (int v = 0; v < 8; ++v) {
      int r = v + 8 * hlf;
      float dg0 = diag[h0 * RNG + r], c0 = Yb[h0 * RNG + r];
      float dg1 = diag[h1 * RNG + r], c1 = Yb[h1 * RNG + r];
      s0 += dg0 * (acc00[v] - c0) * (acc01[v] - c0);
      s1 += dg1 * (acc10[v] - c1) * (acc11[v] - c1);
    }
    s0 += __shfl_xor(s0, 16);   // combine the two r-halves (wave32)
    s1 += __shfl_xor(s1, 16);
    if (hlf == 0) {
      Hout[(size_t)(nodeBase + lane15) * H1 + h0] = s0 + bias0[h0];
      Hout[(size_t)(nodeBase + lane15) * H1 + h1] = s1 + bias0[h1];
    }
  }
}

// ---------------------------------------------------------------------------
// Node GEMM: C[M,J] = A[M,128] @ B[128,J] + bias, B pre-transposed bf16 hi/lo.
// 32 nodes / workgroup, J in {128, 32}.
// ---------------------------------------------------------------------------
#define HKS 136   // LDS K stride for H tile (128 + 8 pad)

__global__ void k_gemm_node(const float* __restrict__ A,
                            const unsigned short* __restrict__ BThi,
                            const unsigned short* __restrict__ BTlo,
                            const float* __restrict__ bias,
                            float* __restrict__ C, int J) {
  __shared__ __align__(16) unsigned short sHi[32 * HKS];
  __shared__ __align__(16) unsigned short sLo[32 * HKS];
  const int base = blockIdx.x * 32;

  for (int i = threadIdx.x; i < 32 * HC; i += 256) {
    int nd = i >> 7, k = i & 127;
    float x = A[(size_t)(base + nd) * HC + k];
    unsigned short h = f2bf(x);
    sHi[nd * HKS + k] = h;
    sLo[nd * HKS + k] = f2bf(x - bf2f(h));
  }
  __syncthreads();

  const int w = threadIdx.x >> 5;
  const int ln = threadIdx.x & 31;
  const int lane15 = ln & 15;
  const int hlf = ln >> 4;
  const int T = 2 * (J >> 4);   // row tiles x col tiles

  for (int t = w; t < T; t += 8) {
    int rt = t & 1, ct = t >> 1;
    v8f acc = {};
#pragma unroll 1
    for (int ks = 0; ks < 4; ++ks) {
      int zz = 0;
      asm volatile("" : "+v"(zz));   // keep loads inside the loop window
      BFrag aHi, aLo, bHi, bLo;
      const unsigned short* ah = sHi + zz + (rt * 16 + lane15) * HKS + ks * 32;
      aHi.q[0] = *(const uint4*)(ah + hlf * 8);
      aHi.q[1] = *(const uint4*)(ah + 16 + hlf * 8);
      const unsigned short* al = sLo + zz + (rt * 16 + lane15) * HKS + ks * 32;
      aLo.q[0] = *(const uint4*)(al + hlf * 8);
      aLo.q[1] = *(const uint4*)(al + 16 + hlf * 8);
      const unsigned short* bh =
          BThi + (size_t)(ct * 16 + lane15) * HC + ks * 32;
      bHi.q[0] = *(const uint4*)(bh + hlf * 8);
      bHi.q[1] = *(const uint4*)(bh + 16 + hlf * 8);
      const unsigned short* bl =
          BTlo + (size_t)(ct * 16 + lane15) * HC + ks * 32;
      bLo.q[0] = *(const uint4*)(bl + hlf * 8);
      bLo.q[1] = *(const uint4*)(bl + 16 + hlf * 8);
      acc = wmma_bf16(aHi, bHi, acc);
      acc = wmma_bf16(aHi, bLo, acc);
      acc = wmma_bf16(aLo, bHi, acc);
    }
    float b = bias[ct * 16 + lane15];
#pragma unroll
    for (int v = 0; v < 8; ++v) {
      int row = base + rt * 16 + v + 8 * hlf;
      C[(size_t)row * J + ct * 16 + lane15] = acc[v] + b;
    }
  }
}

// ---------------------------------------------------------------------------
// GAT edge stages (E' = E + N self loops). One wave (32 lanes) per edge.
// ---------------------------------------------------------------------------
__global__ void k_gat_init(float* __restrict__ amax, float* __restrict__ denom,
                           float* __restrict__ gout) {
  int stride = gridDim.x * blockDim.x;
  for (int i = blockIdx.x * blockDim.x + threadIdx.x; i < NNODE * HC; i += stride) {
    gout[i] = 0.f;
    if (i < NNODE * NHEAD) { amax[i] = -3.0e38f; denom[i] = 0.f; }
  }
}

__global__ void k_edge_alpha(const long long* __restrict__ ei,
                             const float* __restrict__ xl,
                             const float* __restrict__ xr,
                             const float* __restrict__ att,   // [4*32] (layer slice)
                             float* __restrict__ alpha,
                             float* __restrict__ amax) {
  int wid = (blockIdx.x * blockDim.x + threadIdx.x) >> 5;
  int ln = threadIdx.x & 31;
  if (wid >= NEP) return;
  int src, dst;
  if (wid < NEDGE) { src = (int)ei[wid]; dst = (int)ei[NEDGE + wid]; }
  else             { src = dst = wid - NEDGE; }

  float p0, p1, p2, p3;
#define EDGE_H(h, pv)                                                          \
  { float a = xl[(size_t)src * HC + (h) * DHEAD + ln] +                        \
              xr[(size_t)dst * HC + (h) * DHEAD + ln];                         \
    a = a > 0.f ? a : 0.2f * a;                                                \
    pv = a * att[(h) * DHEAD + ln]; }
  EDGE_H(0, p0) EDGE_H(1, p1) EDGE_H(2, p2) EDGE_H(3, p3)
#undef EDGE_H
#pragma unroll
  for (int off = 16; off >= 1; off >>= 1) {
    p0 += __shfl_xor(p0, off);
    p1 += __shfl_xor(p1, off);
    p2 += __shfl_xor(p2, off);
    p3 += __shfl_xor(p3, off);
  }
  if (ln < 4) {
    float v = (ln == 0) ? p0 : (ln == 1) ? p1 : (ln == 2) ? p2 : p3;
    alpha[(size_t)wid * 4 + ln] = v;
    atomic_max_f32(&amax[(size_t)dst * 4 + ln], v);
  }
}

__global__ void k_edge_exp(const long long* __restrict__ ei,
                           float* __restrict__ alpha,     // in: alpha, out: ex
                           const float* __restrict__ amax,
                           float* __restrict__ denom) {
  int tid = blockIdx.x * blockDim.x + threadIdx.x;
  if (tid >= NEP * NHEAD) return;
  int e = tid >> 2, h = tid & 3;
  int dst = (e < NEDGE) ? (int)ei[NEDGE + e] : (e - NEDGE);
  float ex = __expf(alpha[tid] - amax[(size_t)dst * 4 + h]);
  alpha[tid] = ex;
  atomic_add_f32(&denom[(size_t)dst * 4 + h], ex);
}

__global__ void k_edge_scatter(const long long* __restrict__ ei,
                               const float* __restrict__ ex,
                               const float* __restrict__ denom,
                               const float* __restrict__ xl,
                               float* __restrict__ gout) {
  int wid = (blockIdx.x * blockDim.x + threadIdx.x) >> 5;
  int ln = threadIdx.x & 31;
  if (wid >= NEP) return;
  int src, dst;
  if (wid < NEDGE) { src = (int)ei[wid]; dst = (int)ei[NEDGE + wid]; }
  else             { src = dst = wid - NEDGE; }
#pragma unroll
  for (int h = 0; h < NHEAD; ++h) {
    float wgt = ex[(size_t)wid * 4 + h] / denom[(size_t)dst * 4 + h];
    float val = xl[(size_t)src * HC + h * DHEAD + ln] * wgt;
    atomic_add_f32(&gout[(size_t)dst * HC + h * DHEAD + ln], val);
  }
}

__global__ void k_residual(float* __restrict__ H, const float* __restrict__ gout,
                           const float* __restrict__ gb) {   // gb: [128] layer slice
  int i = blockIdx.x * blockDim.x + threadIdx.x;
  if (i >= NNODE * HC) return;
  H[i] += gout[i] + gb[i & (HC - 1)];
}

// ---------------------------------------------------------------------------
// Host side
// ---------------------------------------------------------------------------
extern "C" void kernel_launch(void* const* d_in, const int* in_sizes, int n_in,
                              void* d_out, int out_size, void* d_ws, size_t ws_size,
                              hipStream_t stream) {
  (void)in_sizes; (void)n_in; (void)out_size; (void)ws_size;
  const float*     X     = (const float*)d_in[0];
  const long long* EI    = (const long long*)d_in[1];   // jnp.int64 edge_idx
  const float*     Wbil  = (const float*)d_in[2];
  const float*     diag  = (const float*)d_in[3];
  const float*     bvec  = (const float*)d_in[4];
  const float*     b0    = (const float*)d_in[5];
  const float*     Wl    = (const float*)d_in[6];
  const float*     bl    = (const float*)d_in[7];
  const float*     Wr    = (const float*)d_in[8];
  const float*     br    = (const float*)d_in[9];
  const float*     att   = (const float*)d_in[10];
  const float*     gbias = (const float*)d_in[11];
  const float*     Wlin  = (const float*)d_in[12];
  const float*     blin  = (const float*)d_in[13];
  float* out = (float*)d_out;

  // workspace bump allocator (256B aligned regions)
  size_t off = 0;
  char* base = (char*)d_ws;
  auto alloc = [&](size_t bytes) -> void* {
    void* p = base + off;
    off += (bytes + 255) & ~(size_t)255;
    return p;
  };
  unsigned short* Whi = (unsigned short*)alloc((size_t)H1 * RNG * DIN * 2);
  unsigned short* Wlo = (unsigned short*)alloc((size_t)H1 * RNG * DIN * 2);
  float* Yb    = (float*)alloc((size_t)H1 * RNG * 4);
  float* Hbuf  = (float*)alloc((size_t)NNODE * HC * 4);
  float* xlb   = (float*)alloc((size_t)NNODE * HC * 4);
  float* xrb   = (float*)alloc((size_t)NNODE * HC * 4);
  float* gout  = (float*)alloc((size_t)NNODE * HC * 4);
  float* alpha = (float*)alloc((size_t)NEP * 4 * 4);
  float* amax  = (float*)alloc((size_t)NNODE * 4 * 4);
  float* denom = (float*)alloc((size_t)NNODE * 4 * 4);
  unsigned short *WThi[4], *WTlo[4];
  for (int m = 0; m < 4; ++m) {
    WThi[m] = (unsigned short*)alloc((size_t)HC * HC * 2);
    WTlo[m] = (unsigned short*)alloc((size_t)HC * HC * 2);
  }
  unsigned short* WlinThi = (unsigned short*)alloc((size_t)NCLS * HC * 2);
  unsigned short* WlinTlo = (unsigned short*)alloc((size_t)NCLS * HC * 2);

  // --- prep ---
  k_prep_wsplit<<<(H1 * RNG * DIN + 255) / 256, 256, 0, stream>>>(Wbil, Whi, Wlo);
  k_prep_yb<<<(H1 * RNG + 255) / 256, 256, 0, stream>>>(Wbil, bvec, Yb);
  for (int l = 0; l < 2; ++l) {
    k_prep_ct<<<(HC * HC + 255) / 256, 256, 0, stream>>>(
        Wl + (size_t)l * HC * HC, WThi[l * 2 + 0], WTlo[l * 2 + 0], HC, HC);
    k_prep_ct<<<(HC * HC + 255) / 256, 256, 0, stream>>>(
        Wr + (size_t)l * HC * HC, WThi[l * 2 + 1], WTlo[l * 2 + 1], HC, HC);
  }
  k_prep_ct<<<(HC * NCLS + 255) / 256, 256, 0, stream>>>(Wlin, WlinThi, WlinTlo, HC, NCLS);

  // --- bilinear head: H[N,128] ---
  k_bilinear<<<NNODE / 16, 256, 0, stream>>>(X, Whi, Wlo, diag, Yb, b0, Hbuf);

  // --- GATv2 layers ---
  for (int l = 0; l < 2; ++l) {
    k_gemm_node<<<NNODE / 32, 256, 0, stream>>>(Hbuf, WThi[l * 2 + 0], WTlo[l * 2 + 0],
                                                bl + (size_t)l * HC, xlb, HC);
    k_gemm_node<<<NNODE / 32, 256, 0, stream>>>(Hbuf, WThi[l * 2 + 1], WTlo[l * 2 + 1],
                                                br + (size_t)l * HC, xrb, HC);
    k_gat_init<<<2048, 256, 0, stream>>>(amax, denom, gout);
    k_edge_alpha<<<NEP / 8, 256, 0, stream>>>(EI, xlb, xrb,
                                              att + (size_t)l * NHEAD * DHEAD, alpha, amax);
    k_edge_exp<<<(NEP * 4 + 255) / 256, 256, 0, stream>>>(EI, alpha, amax, denom);
    k_edge_scatter<<<NEP / 8, 256, 0, stream>>>(EI, alpha, denom, xlb, gout);
    k_residual<<<(NNODE * HC + 255) / 256, 256, 0, stream>>>(Hbuf, gout,
                                                             gbias + (size_t)l * HC);
  }

  // --- classifier: out[N,32] ---
  k_gemm_node<<<NNODE / 32, 256, 0, stream>>>(Hbuf, WlinThi, WlinTlo, blin, out, NCLS);
}
